// MultiHeadAttention_14044543058090
// MI455X (gfx1250) — compile-verified, matrix operand order checked by reading, not run on hardware
//
#include <hip/hip_runtime.h>
#include <hip/hip_bf16.h>
#include <cstdint>

typedef __bf16 bf16;
typedef __attribute__((ext_vector_type(16))) __bf16 v16bf;
typedef __attribute__((ext_vector_type(8)))  __bf16 v8bf;
typedef __attribute__((ext_vector_type(4)))  __bf16 v4bf;
typedef __attribute__((ext_vector_type(8)))  float  v8f;

#define Bsz 2
#define Ssz 2048
#define Hsz 1024
#define NHsz 16
#define DKsz 64

// -------- WMMA helpers (CDNA5 wave32, 16x16x32 bf16, f32 accum) --------

__device__ __forceinline__ v8f wmma_bf16(v16bf a, v16bf b, v8f c) {
  // (neg_a, A, neg_b, B, c_mod, C, reuse_a, reuse_b)
  return __builtin_amdgcn_wmma_f32_16x16x32_bf16(false, a, false, b, (short)0, c, false, false);
}

// Lane fragment for a 16x32 (row x K) bf16 tile, row-major in LDS.
// ISA layout: lanes 0-15 hold K=0..7 (v0-3) and K=16..23 (v4-7);
// lanes 16-31 hold K=8..15 and K=24..31; row M = lane%16.
__device__ __forceinline__ v16bf load_frag(const bf16* rowPtr, int lane) {
  const int hi = (lane & 16) ? 8 : 0;
  v8bf c0 = *(const v8bf*)(rowPtr + hi);        // K = hi .. hi+7      (ds_load_b128)
  v8bf c1 = *(const v8bf*)(rowPtr + 16 + hi);   // K = 16+hi .. 16+hi+7
  v16bf f;
#pragma unroll
  for (int i = 0; i < 8; ++i) { f[i] = c0[i]; f[i + 8] = c1[i]; }
  return f;
}

// Stage a 64x32 f32 tile -> bf16 LDS (row stride 40), b128 loads, b64 stores.
__device__ __forceinline__ void stage_A(bf16* dst, const float* src,
                                        int ldSrc, int tid) {
#pragma unroll
  for (int e = tid; e < 64 * 8; e += 256) {
    const int r = e >> 3, c4 = (e & 7) * 4;
    const float4 f = *(const float4*)&src[r * ldSrc + c4];
    v4bf p; p[0] = (bf16)f.x; p[1] = (bf16)f.y; p[2] = (bf16)f.z; p[3] = (bf16)f.w;
    *(v4bf*)&dst[r * 40 + c4] = p;  // 8B aligned -> ds_store_b64
  }
}

// Stage W^T: dst[n][c] = W[c][n] for 32 x 64 slab; b128 global loads.
__device__ __forceinline__ void stage_Bt(bf16* dst, const float* Wslab,
                                         int ldW, int tid) {
#pragma unroll
  for (int e = tid; e < 32 * 16; e += 256) {
    const int c = e >> 4, n4 = (e & 15) * 4;
    const float4 f = *(const float4*)&Wslab[c * ldW + n4];
    dst[(n4 + 0) * 40 + c] = (bf16)f.x;
    dst[(n4 + 1) * 40 + c] = (bf16)f.y;
    dst[(n4 + 2) * 40 + c] = (bf16)f.z;
    dst[(n4 + 3) * 40 + c] = (bf16)f.w;
  }
}

// ---------------- Kernel 1: fused QKV projection (bf16 GEMM) ----------------
// grid (Hsz/64, (B*S)/64, 3), block 256 (8 waves); 64x64 out tile per block.
// Wave w owns subtiles (mi = w/2, n in {2*(w&1), 2*(w&1)+1}).

__global__ __launch_bounds__(256) void proj_qkv_kernel(
    const float* __restrict__ x,
    const float* __restrict__ Wq, const float* __restrict__ bq,
    const float* __restrict__ Wk, const float* __restrict__ bk,
    const float* __restrict__ Wv, const float* __restrict__ bv,
    bf16* __restrict__ Q, bf16* __restrict__ K, bf16* __restrict__ V) {
  const float* W; const float* bias; bf16* out;
  if (blockIdx.z == 0)      { W = Wq; bias = bq; out = Q; }
  else if (blockIdx.z == 1) { W = Wk; bias = bk; out = K; }
  else                      { W = Wv; bias = bv; out = V; }

  __shared__ __align__(16) bf16 As[64 * 40];
  __shared__ __align__(16) bf16 Bs[64 * 40];

  const int tid = threadIdx.x, w = tid >> 5, lane = tid & 31;
  const int mi = w >> 1, nb = (w & 1) * 2;
  const int rowBase = blockIdx.y * 64, colBase = blockIdx.x * 64;

  v8f acc0 = {}, acc1 = {};

  for (int k0 = 0; k0 < Hsz; k0 += 32) {
    stage_A(As, &x[(size_t)rowBase * Hsz + k0], Hsz, tid);
    stage_Bt(Bs, &W[(size_t)k0 * Hsz + colBase], Hsz, tid);
    __syncthreads();

    const bf16* arow  = &As[(mi * 16 + (lane & 15)) * 40];
    const bf16* b0row = &Bs[((nb + 0) * 16 + (lane & 15)) * 40];
    const bf16* b1row = &Bs[((nb + 1) * 16 + (lane & 15)) * 40];
    v16bf a = load_frag(arow, lane);
    acc0 = wmma_bf16(a, load_frag(b0row, lane), acc0);
    acc1 = wmma_bf16(a, load_frag(b1row, lane), acc1);
    __syncthreads();
  }

  // Epilogue: +bias, scatter-store bf16 as [b, head, s, d]
  const int n = lane & 15, mh = (lane & 16) ? 8 : 0;
#pragma unroll
  for (int r = 0; r < 8; ++r) {
    const int gm = rowBase + mi * 16 + mh + r;
    const int bI = gm >> 11, s = gm & (Ssz - 1);
#pragma unroll
    for (int t = 0; t < 2; ++t) {
      const int gn = colBase + (nb + t) * 16 + n;
      const float val = (t == 0 ? acc0[r] : acc1[r]) + bias[gn];
      const int head = gn >> 6, d = gn & (DKsz - 1);
      out[(((bI * NHsz + head) * Ssz) + s) * DKsz + d] = (bf16)val;
    }
  }
}

// ---------------- Kernel 2: flash attention (online softmax) ----------------
// grid (S/64, NH, B), block 256 (8 waves). Block owns a 64-query tile of one
// (b, head); iterates 64-wide KV tiles with running (m, l) row state.

__global__ __launch_bounds__(256) void attention_kernel(
    const bf16* __restrict__ Qbf, const bf16* __restrict__ Kbf,
    const bf16* __restrict__ Vbf,
    const int* __restrict__ mask, const float* __restrict__ alibi,
    float* __restrict__ ctx) {
  __shared__ __align__(16) bf16 Qs[64 * 72];
  __shared__ __align__(16) bf16 Ks[64 * 72];
  __shared__ __align__(16) bf16 Vt[64 * 72];   // transposed: [d][kv]
  __shared__ __align__(16) bf16 Pt[64 * 72];   // probs, bf16
  __shared__ float Sf[64 * 65];                // scores f32
  __shared__ float rowM[64], rowL[64], alphaR[64];
  __shared__ float red[256];                   // 4-way row reductions

  const int tid = threadIdx.x, w = tid >> 5, lane = tid & 31;
  const int mi = w >> 1, nb = (w & 1) * 2;
  const int head = blockIdx.y, batch = blockIdx.z;
  const int qbase = blockIdx.x * 64;

  const bf16* Qh = Qbf + (size_t)(batch * NHsz + head) * Ssz * DKsz;
  const bf16* Kh = Kbf + (size_t)(batch * NHsz + head) * Ssz * DKsz;
  const bf16* Vh = Vbf + (size_t)(batch * NHsz + head) * Ssz * DKsz;

  // Stage Q tile (8B packed copies, coalesced)
#pragma unroll
  for (int e = tid; e < 64 * 16; e += 256) {
    const int r = e >> 4, c = (e & 15) * 4;
    *(unsigned long long*)&Qs[r * 72 + c] =
        *(const unsigned long long*)&Qh[(qbase + r) * DKsz + c];
  }
  if (tid < 64) { rowM[tid] = -INFINITY; rowL[tid] = 0.0f; }

  v8f o0 = {}, o1 = {};
  const int n = lane & 15, mh = (lane & 16) ? 8 : 0;
  const int rRow = tid >> 2, rQ = tid & 3;  // softmax: 4 threads per row

  for (int kt = 0; kt < Ssz / 64; ++kt) {
    const int kvbase = kt * 64;
    // Stage K tile and V transposed
#pragma unroll
    for (int e = tid; e < 64 * 16; e += 256) {
      const int r = e >> 4, c = (e & 15) * 4;
      *(unsigned long long*)&Ks[r * 72 + c] =
          *(const unsigned long long*)&Kh[(kvbase + r) * DKsz + c];
    }
#pragma unroll
    for (int e = tid; e < 64 * 32; e += 256) {
      const int r = e >> 5, c2 = (e & 31) * 2;
      const uint32_t vv = *(const uint32_t*)&Vh[(kvbase + r) * DKsz + c2];
      const bf16* pv = (const bf16*)&vv;
      Vt[(c2 + 0) * 72 + r] = pv[0];
      Vt[(c2 + 1) * 72 + r] = pv[1];
    }
    // Prefetch next tile's alibi/mask stream (global_prefetch_b8)
    if (kt + 1 < Ssz / 64) {
      const int gq = qbase + rRow, gkn = kvbase + 64 + rQ * 16;
      __builtin_prefetch(&alibi[((head * Ssz) + gq) * (size_t)Ssz + gkn], 0, 3);
      __builtin_prefetch(&mask[((batch * Ssz) + gq) * (size_t)Ssz + gkn], 0, 3);
    }
    __syncthreads();

    // S = Q @ K^T  (K tile rows are exactly B^T rows)
    v8f s0 = {}, s1 = {};
    {
      const bf16* qrow  = &Qs[(mi * 16 + (lane & 15)) * 72];
      const bf16* k0row = &Ks[((nb + 0) * 16 + (lane & 15)) * 72];
      const bf16* k1row = &Ks[((nb + 1) * 16 + (lane & 15)) * 72];
#pragma unroll
      for (int kk = 0; kk < DKsz; kk += 32) {
        v16bf a = load_frag(qrow + kk, lane);
        s0 = wmma_bf16(a, load_frag(k0row + kk, lane), s0);
        s1 = wmma_bf16(a, load_frag(k1row + kk, lane), s1);
      }
    }

    // scale + alibi + mask, spill scores to LDS
#pragma unroll
    for (int r = 0; r < 8; ++r) {
      const int lm = mi * 16 + mh + r;
      const int gq = qbase + lm;
#pragma unroll
      for (int t = 0; t < 2; ++t) {
        const int ln = (nb + t) * 16 + n;
        const int gk = kvbase + ln;
        float sc = (t == 0 ? s0[r] : s1[r]) * 0.125f +
                   alibi[((head * Ssz) + gq) * (size_t)Ssz + gk];
        if (mask[((batch * Ssz) + gq) * (size_t)Ssz + gk] == 0) sc = -1e9f;
        Sf[lm * 65 + ln] = sc;
      }
    }
    __syncthreads();

    // --- Online softmax, 4 threads per row ---
    {  // pass A: partial row max
      float mt = -INFINITY;
#pragma unroll
      for (int j = 0; j < 16; ++j) mt = fmaxf(mt, Sf[rRow * 65 + rQ * 16 + j]);
      red[rRow * 4 + rQ] = mt;
    }
    __syncthreads();
    if (tid < 64) {  // pass B: combine, compute alpha
      const float mOld = rowM[tid];
      float mt = fmaxf(fmaxf(red[tid * 4 + 0], red[tid * 4 + 1]),
                       fmaxf(red[tid * 4 + 2], red[tid * 4 + 3]));
      const float mNew = fmaxf(mOld, mt);
      alphaR[tid] = __expf(mOld - mNew);  // first iter: exp(-inf)=0
      rowM[tid] = mNew;
    }
    __syncthreads();
    {  // pass C: exponentiate + bf16 probs + partial sums
      const float mNew = rowM[rRow];
      float s = 0.0f;
#pragma unroll
      for (int j = 0; j < 16; ++j) {
        const float p = __expf(Sf[rRow * 65 + rQ * 16 + j] - mNew);
        Pt[rRow * 72 + rQ * 16 + j] = (bf16)p;
        s += p;
      }
      red[rRow * 4 + rQ] = s;
    }
    __syncthreads();
    if (tid < 64) {  // pass D: update running denominator
      rowL[tid] = rowL[tid] * alphaR[tid] +
                  (red[tid * 4 + 0] + red[tid * 4 + 1] +
                   red[tid * 4 + 2] + red[tid * 4 + 3]);
    }

    // Rescale running O, then O += P @ V
    float al[8];
#pragma unroll
    for (int r = 0; r < 8; ++r) al[r] = alphaR[mi * 16 + mh + r];
#pragma unroll
    for (int r = 0; r < 8; ++r) { o0[r] *= al[r]; o1[r] *= al[r]; }
    {
      const bf16* prow  = &Pt[(mi * 16 + (lane & 15)) * 72];
      const bf16* v0row = &Vt[((nb + 0) * 16 + (lane & 15)) * 72];
      const bf16* v1row = &Vt[((nb + 1) * 16 + (lane & 15)) * 72];
#pragma unroll
      for (int kk = 0; kk < 64; kk += 32) {
        v16bf a = load_frag(prow + kk, lane);
        o0 = wmma_bf16(a, load_frag(v0row + kk, lane), o0);
        o1 = wmma_bf16(a, load_frag(v1row + kk, lane), o1);
      }
    }
    __syncthreads();
  }

  // Final normalize + store ctx as [b, s, head*64 + d] (f32)
#pragma unroll
  for (int r = 0; r < 8; ++r) {
    const int lm = mi * 16 + mh + r;
    const float invl = 1.0f / rowL[lm];
    const int gq = qbase + lm;
#pragma unroll
    for (int t = 0; t < 2; ++t) {
      const int gn = head * DKsz + (nb + t) * 16 + n;
      ctx[((size_t)(batch * Ssz) + gq) * Hsz + gn] = (t == 0 ? o0[r] : o1[r]) * invl;
    }
  }
}

// ---------------- Kernel 3: output projection (f32 out) ----------------

__global__ __launch_bounds__(256) void proj_out_kernel(
    const float* __restrict__ A, const float* __restrict__ W,
    const float* __restrict__ bias, float* __restrict__ out) {
  __shared__ __align__(16) bf16 As[64 * 40];
  __shared__ __align__(16) bf16 Bs[64 * 40];

  const int tid = threadIdx.x, w = tid >> 5, lane = tid & 31;
  const int mi = w >> 1, nb = (w & 1) * 2;
  const int rowBase = blockIdx.y * 64, colBase = blockIdx.x * 64;

  v8f acc0 = {}, acc1 = {};
  for (int k0 = 0; k0 < Hsz; k0 += 32) {
    stage_A(As, &A[(size_t)rowBase * Hsz + k0], Hsz, tid);
    stage_Bt(Bs, &W[(size_t)k0 * Hsz + colBase], Hsz, tid);
    __syncthreads();
    const bf16* arow  = &As[(mi * 16 + (lane & 15)) * 40];
    const bf16* b0row = &Bs[((nb + 0) * 16 + (lane & 15)) * 40];
    const bf16* b1row = &Bs[((nb + 1) * 16 + (lane & 15)) * 40];
    v16bf a = load_frag(arow, lane);
    acc0 = wmma_bf16(a, load_frag(b0row, lane), acc0);
    acc1 = wmma_bf16(a, load_frag(b1row, lane), acc1);
    __syncthreads();
  }

  const int n = lane & 15, mh = (lane & 16) ? 8 : 0;
#pragma unroll
  for (int r = 0; r < 8; ++r) {
    const int gm = rowBase + mi * 16 + mh + r;
#pragma unroll
    for (int t = 0; t < 2; ++t) {
      const int gn = colBase + (nb + t) * 16 + n;
      out[(size_t)gm * Hsz + gn] = (t == 0 ? acc0[r] : acc1[r]) + bias[gn];
    }
  }
}

// ---------------- Launch ----------------

extern "C" void kernel_launch(void* const* d_in, const int* in_sizes, int n_in,
                              void* d_out, int out_size, void* d_ws, size_t ws_size,
                              hipStream_t stream) {
  (void)in_sizes; (void)n_in; (void)out_size; (void)ws_size;
  const float* x     = (const float*)d_in[0];
  const int*   mask  = (const int*)  d_in[1];
  const float* alibi = (const float*)d_in[2];
  const float* Wq    = (const float*)d_in[3];
  const float* bq    = (const float*)d_in[4];
  const float* Wk    = (const float*)d_in[5];
  const float* bk    = (const float*)d_in[6];
  const float* Wv    = (const float*)d_in[7];
  const float* bv    = (const float*)d_in[8];
  const float* Wo    = (const float*)d_in[9];
  const float* bo    = (const float*)d_in[10];

  const size_t QKV_BYTES = (size_t)Bsz * NHsz * Ssz * DKsz * sizeof(bf16);  // 8 MB each
  char* ws = (char*)d_ws;
  bf16*  Q   = (bf16*)(ws);
  bf16*  K   = (bf16*)(ws + QKV_BYTES);
  bf16*  V   = (bf16*)(ws + 2 * QKV_BYTES);
  float* ctx = (float*)(ws + 3 * QKV_BYTES);

  proj_qkv_kernel<<<dim3(Hsz / 64, (Bsz * Ssz) / 64, 3), 256, 0, stream>>>(
      x, Wq, bq, Wk, bk, Wv, bv, Q, K, V);
  attention_kernel<<<dim3(Ssz / 64, NHsz, Bsz), 256, 0, stream>>>(
      Q, K, V, mask, alibi, ctx);
  proj_out_kernel<<<dim3(Hsz / 64, (Bsz * Ssz) / 64, 1), 256, 0, stream>>>(
      ctx, Wo, bo, (float*)d_out);
}
